// SparseMoE_64458869178976
// MI455X (gfx1250) — matrix operand order repaired and checked by reading, not compiled, not run on hardware
//
#include <hip/hip_runtime.h>
#include <hip/hip_bf16.h>
#include <cstdint>
#include <math.h>

// Problem constants (B,T,D,E,H = 4,2048,1024,8,4096; k=2)
#define N_TOK 8192
#define DDIM  1024
#define EEXP  8
#define HDIM  4096

// Tile config: block = 256 threads = 8 waves. Block tile 64(M) x 64(N), K=32.
// Waves form a 4(M) x 2(N-pair) grid; each wave owns two 16x16 N-subtiles.
#define BM 64
#define BN 64
#define BK 32
#define KPAD 8   // LDS row padding (elements); row stride 40*2 = 80B (16B mult.)

typedef __attribute__((ext_vector_type(16))) __bf16 v16bf;
typedef __attribute__((ext_vector_type(8)))  __bf16 v8bf;
typedef __attribute__((ext_vector_type(8)))  float  v8f;

// Matches the async builtin's parameter type exactly (from round-1 diagnostic):
// '__attribute__((__vector_size__(4 * sizeof(int)))) int *'
typedef int v4i_vs __attribute__((vector_size(16)));
typedef __attribute__((address_space(1))) v4i_vs* gptr_v4i;
typedef __attribute__((address_space(3))) v4i_vs* lptr_v4i;

#if defined(__has_builtin)
#if __has_builtin(__builtin_amdgcn_global_load_async_to_lds_b128)
#define USE_ASYNC_LDS 1
#endif
#endif

#ifdef USE_ASYNC_LDS
#define ASYNC_CP16(gsrc, ldst)                                               \
  __builtin_amdgcn_global_load_async_to_lds_b128(                            \
      (gptr_v4i)(uintptr_t)(gsrc), (lptr_v4i)(unsigned)(uintptr_t)(ldst), 0, 0)
#if __has_builtin(__builtin_amdgcn_s_wait_asynccnt)
#define WAIT_ASYNC(n) __builtin_amdgcn_s_wait_asynccnt(n)
#else
#define WAIT_ASYNC(n) asm volatile("s_wait_asynccnt %0" ::"i"(n) : "memory")
#endif
#endif

// Build a 16-element bf16 fragment from two contiguous 16B LDS runs.
#define FRAG16(base, off)                                                     \
  __builtin_shufflevector(*(const v8bf*)&(base)[(off)],                       \
                          *(const v8bf*)&(base)[(off) + 8],                   \
                          0, 1, 2, 3, 4, 5, 6, 7, 8, 9, 10, 11, 12, 13, 14, 15)

// ---------------------------------------------------------------------------
// Gating: one wave per token; top-2 + softmax. Deterministic (no atomics).
// ---------------------------------------------------------------------------
__global__ __launch_bounds__(256)
void moe_gate(const float* __restrict__ x,
              const float* __restrict__ gw,
              const float* __restrict__ gb,
              int* __restrict__ idx_pair,
              float* __restrict__ prob_pair) {
  const int tok  = (int)((blockIdx.x * blockDim.x + threadIdx.x) >> 5);
  const int lane = (int)(threadIdx.x & 31);
  if (tok >= N_TOK) return;

  const float* xr = x + (size_t)tok * DDIM;
  float acc[EEXP];
#pragma unroll
  for (int e = 0; e < EEXP; ++e) acc[e] = 0.f;

  for (int d = lane; d < DDIM; d += 32) {
    const float xv = xr[d];
    const float* g = gw + (size_t)d * EEXP;
#pragma unroll
    for (int e = 0; e < EEXP; ++e) acc[e] += xv * g[e];
  }
#pragma unroll
  for (int e = 0; e < EEXP; ++e) {
#pragma unroll
    for (int off = 16; off > 0; off >>= 1)
      acc[e] += __shfl_xor(acc[e], off, 32);
  }

  if (lane == 0) {
    float lg[EEXP];
#pragma unroll
    for (int e = 0; e < EEXP; ++e) lg[e] = acc[e] + gb[e];
    int i0 = 0;
#pragma unroll
    for (int e = 1; e < EEXP; ++e) if (lg[e] > lg[i0]) i0 = e;
    int i1 = (i0 == 0) ? 1 : 0;
#pragma unroll
    for (int e = 0; e < EEXP; ++e) if (e != i0 && lg[e] > lg[i1]) i1 = e;
    const float p1 = expf(lg[i1] - lg[i0]);
    const float s  = 1.f + p1;
    idx_pair[2 * tok]      = i0;
    idx_pair[2 * tok + 1]  = i1;
    prob_pair[2 * tok]     = 1.f / s;
    prob_pair[2 * tok + 1] = p1 / s;
  }
}

// ---------------------------------------------------------------------------
// Deterministic per-expert token lists (token-order scan; 1 thread / expert).
// ---------------------------------------------------------------------------
__global__ void moe_build_lists(const int* __restrict__ idx_pair,
                                const float* __restrict__ prob_pair,
                                int* __restrict__ cnt,
                                int* __restrict__ tok_list,
                                float* __restrict__ w_list) {
  const int e = (int)threadIdx.x;
  if (e >= EEXP) return;
  int c = 0;
  for (int t = 0; t < N_TOK; ++t) {
#pragma unroll
    for (int s = 0; s < 2; ++s) {
      if (idx_pair[2 * t + s] == e) {
        tok_list[(size_t)e * N_TOK + c] = t;
        w_list[(size_t)e * N_TOK + c]   = prob_pair[2 * t + s];
        ++c;
      }
    }
  }
  cnt[e] = c;
}

// ---------------------------------------------------------------------------
// x f32 [N,D] -> bf16 [N,D] (once per call; 4 elements / thread).
// ---------------------------------------------------------------------------
__global__ __launch_bounds__(256)
void moe_cvt_x(const float* __restrict__ x, __bf16* __restrict__ xb) {
  const size_t i = ((size_t)blockIdx.x * blockDim.x + threadIdx.x) * 4;
  if (i + 3 < (size_t)N_TOK * DDIM) {
    const float4 v = *(const float4*)(x + i);
    xb[i + 0] = (__bf16)v.x;
    xb[i + 1] = (__bf16)v.y;
    xb[i + 2] = (__bf16)v.z;
    xb[i + 3] = (__bf16)v.w;
  }
}

// ---------------------------------------------------------------------------
// Pack: w [R][C] f32 -> wt [C][R] bf16 (K-major panels for WMMA B-fragments).
// 32x32 LDS tile transpose, coalesced on both sides.
// ---------------------------------------------------------------------------
__global__ __launch_bounds__(256)
void moe_pack_wt(const float* __restrict__ w, __bf16* __restrict__ wt,
                 int R, int C) {
  __shared__ float tile[32][33];
  const int c0 = (int)blockIdx.x * 32, r0 = (int)blockIdx.y * 32;
  const int tc = (int)threadIdx.x & 31, tr = (int)threadIdx.x >> 5;  // tr 0..7
#pragma unroll
  for (int i = 0; i < 4; ++i)
    tile[tr + i * 8][tc] = w[(size_t)(r0 + tr + i * 8) * C + (c0 + tc)];
  __syncthreads();
#pragma unroll
  for (int i = 0; i < 4; ++i)
    wt[(size_t)(c0 + tr + i * 8) * R + (r0 + tc)] = (__bf16)tile[tc][tr + i * 8];
}

// ---------------------------------------------------------------------------
// GEMM1 (per expert): h[slot,:] = silu(x W1 + b1) * (x W2 + b2), bf16 WMMA.
// 64x64 block tile: every panel is 256 x 16B chunks -> exactly one async
// instruction per thread per panel -> unconditional s_wait_asynccnt.
// ---------------------------------------------------------------------------
__global__ __launch_bounds__(256)
void moe_gemm1(const __bf16* __restrict__ xb,
               const __bf16* __restrict__ w1t,   // [H][D] bf16, K-major
               const float*  __restrict__ b1e,
               const __bf16* __restrict__ w2t,   // [H][D] bf16, K-major
               const float*  __restrict__ b2e,
               const int* __restrict__ tok_e, const int* __restrict__ cnt_e,
               __bf16* __restrict__ h_buf) {
  const int count = *cnt_e;
  const int row0  = (int)blockIdx.y * BM;
  if (row0 >= count) return;
  const int n0 = (int)blockIdx.x * BN;

  __shared__ __align__(16) __bf16 Xs[2][BM][BK + KPAD];
  __shared__ __align__(16) __bf16 W1s[2][BN][BK + KPAD];
  __shared__ __align__(16) __bf16 W2s[2][BN][BK + KPAD];
  __shared__ int toks[BM];

  const int tid = (int)threadIdx.x;
  if (tid < BM) {
    const int slot = row0 + tid;
    toks[tid] = (slot < count) ? tok_e[slot] : tok_e[count - 1];
  }
  __syncthreads();

  // Staging coordinates: 16B chunk (tid&3), row (tid>>2); 256 chunks / panel.
  const int rA  = tid >> 2;          // 0..63
  const int chA = (tid & 3) * 8;
  const __bf16* gX  = xb  + (size_t)toks[rA] * DDIM + chA;
  const __bf16* gW1 = w1t + (size_t)(n0 + rA) * DDIM + chA;
  const __bf16* gW2 = w2t + (size_t)(n0 + rA) * DDIM + chA;

  const int wv = tid >> 5, lane = tid & 31;
  const int wm  = wv & 3;            // M subtile 0..3
  const int wnp = (wv >> 2) * 2;     // N subtile pair base: 0 or 2
  const int half = lane >> 4, ln = lane & 15;
  const int am  = wm * 16 + ln;      // A row in block tile
  const int bn0 = wnp * 16 + ln;     // B rows (output columns) of the pair
  const int bn1 = bn0 + 16;

  auto stage = [&](int buf, int k0) {
#ifdef USE_ASYNC_LDS
    ASYNC_CP16(gW1 + k0, &W1s[buf][rA][chA]);
    ASYNC_CP16(gW2 + k0, &W2s[buf][rA][chA]);
    ASYNC_CP16(gX  + k0, &Xs[buf][rA][chA]);
#else
#pragma unroll
    for (int i = 0; i < (BN * BK) / 256; ++i) {
      const int q = tid + i * 256, r = q >> 5, c = q & 31;
      W1s[buf][r][c] = w1t[(size_t)(n0 + r) * DDIM + k0 + c];
      W2s[buf][r][c] = w2t[(size_t)(n0 + r) * DDIM + k0 + c];
      Xs[buf][r][c]  = xb[(size_t)toks[r] * DDIM + k0 + c];
    }
#endif
  };

  v8f acc1[2] = {{}, {}}, acc2[2] = {{}, {}};
  stage(0, 0);

  for (int k0 = 0; k0 < DDIM; k0 += BK) {
    const int  cur  = (k0 >> 5) & 1;
    const bool more = (k0 + BK) < DDIM;
    if (more) stage(cur ^ 1, k0 + BK);
#ifdef USE_ASYNC_LDS
    // 3 async issues per wave per tile, in-order completion: waiting down to
    // 3 outstanding guarantees the current buffer has landed (0 on last iter).
    if (more) WAIT_ASYNC(3); else WAIT_ASYNC(0);
#endif
    __syncthreads();   // current tile visible to all waves

    const v16bf a = __builtin_shufflevector(
        *(const v8bf*)&Xs[cur][am][half * 16],
        *(const v8bf*)&Xs[cur][am][half * 16 + 8],
        0, 1, 2, 3, 4, 5, 6, 7, 8, 9, 10, 11, 12, 13, 14, 15);
    const v16bf b10 = FRAG16(W1s[cur][bn0], 16 * half);
    const v16bf b11 = FRAG16(W1s[cur][bn1], 16 * half);
    const v16bf b20 = FRAG16(W2s[cur][bn0], 16 * half);
    const v16bf b21 = FRAG16(W2s[cur][bn1], 16 * half);

    acc1[0] = __builtin_amdgcn_wmma_f32_16x16x32_bf16(false, a, false, b10,
                                                      (short)0, acc1[0], false, false);
    acc1[1] = __builtin_amdgcn_wmma_f32_16x16x32_bf16(false, a, false, b11,
                                                      (short)0, acc1[1], false, false);
    acc2[0] = __builtin_amdgcn_wmma_f32_16x16x32_bf16(false, a, false, b20,
                                                      (short)0, acc2[0], false, false);
    acc2[1] = __builtin_amdgcn_wmma_f32_16x16x32_bf16(false, a, false, b21,
                                                      (short)0, acc2[1], false, false);
    __syncthreads();   // everyone done with 'cur' before it is refilled
  }

#pragma unroll
  for (int g = 0; g < 2; ++g) {
    const int col = n0 + (wnp + g) * 16 + ln;
    const float bias1 = b1e[col], bias2 = b2e[col];
#pragma unroll
    for (int r = 0; r < 8; ++r) {    // C/D: M = r + 8*half, N = lane%16
      const int slot = row0 + wm * 16 + half * 8 + r;
      if (slot < count) {
        const float av = acc1[g][r] + bias1;
        const float bv = acc2[g][r] + bias2;
        const float hv = (av / (1.f + expf(-av))) * bv;   // silu(a) * b
        h_buf[(size_t)slot * HDIM + col] = (__bf16)hv;
      }
    }
  }
}

// ---------------------------------------------------------------------------
// GEMM2 (per expert): out[tok,:] += gate * (h Wp + bp). Stream-serialized
// expert kernels + unique token per list slot -> plain += is deterministic.
// ---------------------------------------------------------------------------
__global__ __launch_bounds__(256)
void moe_gemm2(const __bf16* __restrict__ h_buf,   // [N,H] bf16
               const __bf16* __restrict__ wpt,     // [D][H] bf16, K-major
               const float*  __restrict__ bpe,
               const int* __restrict__ tok_e, const float* __restrict__ wgt_e,
               const int* __restrict__ cnt_e,
               float* __restrict__ out) {
  const int count = *cnt_e;
  const int row0  = (int)blockIdx.y * BM;
  if (row0 >= count) return;
  const int n0 = (int)blockIdx.x * BN;

  __shared__ __align__(16) __bf16 Hs[2][BM][BK + KPAD];
  __shared__ __align__(16) __bf16 Ws[2][BN][BK + KPAD];

  const int tid = (int)threadIdx.x;
  const int rA  = tid >> 2;
  const int chA = (tid & 3) * 8;
  const __bf16* gH = h_buf + (size_t)(row0 + rA) * HDIM + chA;  // rows < N_TOK
  const __bf16* gW = wpt + (size_t)(n0 + rA) * HDIM + chA;

  const int wv = tid >> 5, lane = tid & 31;
  const int wm  = wv & 3;
  const int wnp = (wv >> 2) * 2;
  const int half = lane >> 4, ln = lane & 15;
  const int am  = wm * 16 + ln;
  const int bn0 = wnp * 16 + ln;
  const int bn1 = bn0 + 16;

  auto stage = [&](int buf, int k0) {
#ifdef USE_ASYNC_LDS
    ASYNC_CP16(gW + k0, &Ws[buf][rA][chA]);
    ASYNC_CP16(gH + k0, &Hs[buf][rA][chA]);
#else
#pragma unroll
    for (int i = 0; i < (BN * BK) / 256; ++i) {
      const int q = tid + i * 256, r = q >> 5, c = q & 31;
      Ws[buf][r][c] = wpt[(size_t)(n0 + r) * HDIM + k0 + c];
      Hs[buf][r][c] = h_buf[(size_t)(row0 + r) * HDIM + k0 + c];
    }
#endif
  };

  v8f acc[2] = {{}, {}};
  stage(0, 0);

  for (int k0 = 0; k0 < HDIM; k0 += BK) {
    const int  cur  = (k0 >> 5) & 1;
    const bool more = (k0 + BK) < HDIM;
    if (more) stage(cur ^ 1, k0 + BK);
#ifdef USE_ASYNC_LDS
    if (more) WAIT_ASYNC(2); else WAIT_ASYNC(0);
#endif
    __syncthreads();

    const v16bf a = __builtin_shufflevector(
        *(const v8bf*)&Hs[cur][am][half * 16],
        *(const v8bf*)&Hs[cur][am][half * 16 + 8],
        0, 1, 2, 3, 4, 5, 6, 7, 8, 9, 10, 11, 12, 13, 14, 15);
    const v16bf b0 = FRAG16(Ws[cur][bn0], 16 * half);
    const v16bf b1 = FRAG16(Ws[cur][bn1], 16 * half);

    acc[0] = __builtin_amdgcn_wmma_f32_16x16x32_bf16(false, a, false, b0,
                                                     (short)0, acc[0], false, false);
    acc[1] = __builtin_amdgcn_wmma_f32_16x16x32_bf16(false, a, false, b1,
                                                     (short)0, acc[1], false, false);
    __syncthreads();
  }

#pragma unroll
  for (int g = 0; g < 2; ++g) {
    const int col = n0 + (wnp + g) * 16 + ln;
    const float bias = bpe[col];
#pragma unroll
    for (int r = 0; r < 8; ++r) {
      const int slot = row0 + wm * 16 + half * 8 + r;
      if (slot < count) {
        const int tok  = tok_e[slot];
        const float wg = wgt_e[slot];
        out[(size_t)tok * DDIM + col] += wg * (acc[g][r] + bias);
      }
    }
  }
}

// ---------------------------------------------------------------------------
// aux = sum((counts/total - 1/E)^2)
// ---------------------------------------------------------------------------
__global__ void moe_aux(const int* __restrict__ cnt, float* __restrict__ aux_out) {
  if (threadIdx.x == 0 && blockIdx.x == 0) {
    float tot = 0.f;
    for (int e = 0; e < EEXP; ++e) tot += (float)cnt[e];
    float a = 0.f;
    for (int e = 0; e < EEXP; ++e) {
      const float f = (float)cnt[e] / tot - 1.f / (float)EEXP;
      a += f * f;
    }
    *aux_out = a;
  }
}

// ---------------------------------------------------------------------------
// Launch
// ---------------------------------------------------------------------------
extern "C" void kernel_launch(void* const* d_in, const int* in_sizes, int n_in,
                              void* d_out, int out_size, void* d_ws, size_t ws_size,
                              hipStream_t stream) {
  const float* x  = (const float*)d_in[0];
  const float* gw = (const float*)d_in[1];
  const float* gb = (const float*)d_in[2];
  const float* w1 = (const float*)d_in[3];
  const float* b1 = (const float*)d_in[4];
  const float* w2 = (const float*)d_in[5];
  const float* b2 = (const float*)d_in[6];
  const float* wp = (const float*)d_in[7];
  const float* bp = (const float*)d_in[8];
  // d_in[9] = k == 2 (baked in)

  float* out = (float*)d_out;  // [N,D] flat + aux scalar at index N*D

  // Workspace layout (16B/256B-aligned offsets):
  char*   ws        = (char*)d_ws;
  int*    cnt       = (int*)ws;                                    // E ints
  int*    idx_pair  = (int*)(ws + 256);                            // 2N ints
  float*  prob_pair = (float*)(ws + 256 + 2 * N_TOK * 4);          // 2N floats
  int*    tok_list  = (int*)(ws + 256 + 4 * N_TOK * 4);            // E*N ints
  float*  w_list    = (float*)(ws + 256 + 4 * N_TOK * 4 + EEXP * N_TOK * 4);
  size_t  off       = 256 + 4 * (size_t)N_TOK * 4 + 2 * (size_t)EEXP * N_TOK * 4;
  __bf16* xb   = (__bf16*)(ws + off);                    off += (size_t)N_TOK * DDIM * 2;
  __bf16* w1t  = (__bf16*)(ws + off);                    off += (size_t)DDIM * HDIM * 2;
  __bf16* w2t  = (__bf16*)(ws + off);                    off += (size_t)DDIM * HDIM * 2;
  __bf16* wpt  = (__bf16*)(ws + off);                    off += (size_t)DDIM * HDIM * 2;
  __bf16* h_buf = (__bf16*)(ws + off);                   // N*H bf16

  // out accumulates across expert kernels -> must be zeroed every call.
  (void)hipMemsetAsync(d_out, 0, (size_t)out_size * sizeof(float), stream);

  moe_gate<<<N_TOK / 8, 256, 0, stream>>>(x, gw, gb, idx_pair, prob_pair);
  moe_build_lists<<<1, EEXP, 0, stream>>>(idx_pair, prob_pair, cnt, tok_list, w_list);
  moe_cvt_x<<<(N_TOK * DDIM / 4) / 256, 256, 0, stream>>>(x, xb);

  for (int e = 0; e < EEXP; ++e) {
    // Pack this expert's weights into bf16 K-major panels (hits L2 on reuse).
    moe_pack_wt<<<dim3(HDIM / 32, DDIM / 32), 256, 0, stream>>>(
        w1 + (size_t)e * DDIM * HDIM, w1t, DDIM, HDIM);
    moe_pack_wt<<<dim3(HDIM / 32, DDIM / 32), 256, 0, stream>>>(
        w2 + (size_t)e * DDIM * HDIM, w2t, DDIM, HDIM);
    moe_gemm1<<<dim3(HDIM / BN, N_TOK / BM), 256, 0, stream>>>(
        xb, w1t, b1 + (size_t)e * HDIM, w2t, b2 + (size_t)e * HDIM,
        tok_list + (size_t)e * N_TOK, cnt + e, h_buf);
    moe_pack_wt<<<dim3(DDIM / 32, HDIM / 32), 256, 0, stream>>>(
        wp + (size_t)e * HDIM * DDIM, wpt, HDIM, DDIM);
    moe_gemm2<<<dim3(DDIM / BN, N_TOK / BM), 256, 0, stream>>>(
        h_buf, wpt, bp + (size_t)e * DDIM,
        tok_list + (size_t)e * N_TOK, w_list + (size_t)e * N_TOK,
        cnt + e, out);
  }

  moe_aux<<<1, 32, 0, stream>>>(cnt, out + (size_t)N_TOK * DDIM);
}
